// EncoderLayer_27951647162538
// MI455X (gfx1250) — matrix operand order repaired and checked by reading, not compiled, not run on hardware
//
#include <hip/hip_runtime.h>
#include <hip/hip_bf16.h>

// ---------------------------------------------------------------------------
// Types for CDNA5 WMMA (wave32): D = A(16x32 bf16) * B(32x16 bf16) + C(16x16 f32)
// ---------------------------------------------------------------------------
typedef __attribute__((ext_vector_type(16))) __bf16 v16bf;
typedef __attribute__((ext_vector_type(8)))  float  v8f;
typedef __attribute__((ext_vector_type(4)))  int    v4i;

union FragBF {
    v16bf v;
    uint4 q[2];
};

static __device__ inline v8f wmma_bf16(const FragBF& a, const FragBF& b, v8f c) {
    // emits v_wmma_f32_16x16x32_bf16
    return __builtin_amdgcn_wmma_f32_16x16x32_bf16(
        /*neg_a=*/false, a.v, /*neg_b=*/false, b.v,
        /*c_mod=*/(short)0, c, /*reuse_a=*/false, /*reuse_b=*/false);
}

// gfx1250 async global->LDS staging (ASYNCcnt) with safe fallback
#if defined(__gfx1250__) && __has_builtin(__builtin_amdgcn_global_load_async_to_lds_b128)
#define ASYNC_LDS 1
#else
#define ASYNC_LDS 0
#endif

static __device__ inline void ld_tile16(const __bf16* __restrict__ g, __bf16* l) {
#if ASYNC_LDS
    __builtin_amdgcn_global_load_async_to_lds_b128(
        (__attribute__((address_space(1))) v4i*)g,
        (__attribute__((address_space(3))) v4i*)l, 0, 0);
#else
    *(uint4*)l = *(const uint4*)g;
#endif
}

static __device__ inline void async_wait0() {
#if ASYNC_LDS
#if __has_builtin(__builtin_amdgcn_s_wait_asynccnt)
    __builtin_amdgcn_s_wait_asynccnt(0);
#else
    asm volatile("s_wait_asynccnt 0" ::: "memory");
#endif
#endif
}

#define D_MODEL 768
#define D_FF    3072
#define SEQ     2048
#define NHEAD   12
#define DKH     64

// ---------------------------------------------------------------------------
// fp32[K][N] -> bf16[N][K] tiled transpose-convert (weights, one-time)
// block = 256 (32x8), 32x32 tile, coalesced on both sides
// ---------------------------------------------------------------------------
__global__ __launch_bounds__(256)
void transpose_cvt_kernel(const float* __restrict__ src, __bf16* __restrict__ dst,
                          int K, int N) {
    __shared__ float t[32][33];
    const int k0 = blockIdx.x * 32, n0 = blockIdx.y * 32;
    const int tx = threadIdx.x & 31, ty = threadIdx.x >> 5;
#pragma unroll
    for (int i = 0; i < 4; ++i) {
        const int kr = ty + i * 8;
        t[kr][tx] = src[(long)(k0 + kr) * N + n0 + tx];
    }
    __syncthreads();
#pragma unroll
    for (int i = 0; i < 4; ++i) {
        const int nr = ty + i * 8;
        dst[(long)(n0 + nr) * K + k0 + tx] = (__bf16)t[tx][nr];
    }
}

// ---------------------------------------------------------------------------
// LayerNorm (reference variant: scalar alpha/bias, ddof=1, denom = eps+std)
// one 256-thread block per row of 768 floats; output bf16
// ---------------------------------------------------------------------------
__global__ __launch_bounds__(256)
void ln_bf16_kernel(const float* __restrict__ x, const float* __restrict__ alpha,
                    const float* __restrict__ beta, __bf16* __restrict__ y) {
    __shared__ float red[256];
    const long row = blockIdx.x;
    const float* xr = x + row * (long)D_MODEL;
    const int tid = threadIdx.x;

    float v[3];
    float s = 0.f;
#pragma unroll
    for (int i = 0; i < 3; ++i) { v[i] = xr[tid + i * 256]; s += v[i]; }
    red[tid] = s;
    __syncthreads();
    for (int off = 128; off > 0; off >>= 1) {
        if (tid < off) red[tid] += red[tid + off];
        __syncthreads();
    }
    const float mean = red[0] * (1.0f / D_MODEL);
    __syncthreads();

    float ss = 0.f;
#pragma unroll
    for (int i = 0; i < 3; ++i) { float d = v[i] - mean; ss += d * d; }
    red[tid] = ss;
    __syncthreads();
    for (int off = 128; off > 0; off >>= 1) {
        if (tid < off) red[tid] += red[tid + off];
        __syncthreads();
    }
    const float var = red[0] * (1.0f / (D_MODEL - 1));
    const float scl = alpha[0] / (1e-6f + sqrtf(var));
    const float b0 = beta[0];
#pragma unroll
    for (int i = 0; i < 3; ++i)
        y[row * (long)D_MODEL + tid + i * 256] = (__bf16)((v[i] - mean) * scl + b0);
}

// ---------------------------------------------------------------------------
// bf16 WMMA GEMM with pre-transposed weights:
//   out[M,N] = epi(A[M,K] @ Wt[N,K]^T + bias)
// BLOCK 128x128, BK=32, 256 threads = 8 waves (4x2), wave tile 32x64 -> 8 wmma.
// epi: 0 = store bf16, 1 = relu -> bf16, 2 = residual + f32 store
// ---------------------------------------------------------------------------
#define LDT_S 40   // padded LDS stride (bf16 elems), 80B keeps 16B row alignment

__global__ __launch_bounds__(256)
void gemm_bf16_wmma_kernel(const __bf16* __restrict__ A, const __bf16* __restrict__ Wt,
                           const float* __restrict__ bias, void* __restrict__ out,
                           const float* __restrict__ resid,
                           int M, int N, int K, int epi) {
    __shared__ __attribute__((aligned(16))) __bf16 As[128 * LDT_S];  // [m][k]
    __shared__ __attribute__((aligned(16))) __bf16 Bs[128 * LDT_S];  // [n][k]

    const int tid = threadIdx.x;
    const int lane = tid & 31, w = tid >> 5;
    const int half = lane >> 4, li = lane & 15;
    const int wm = w >> 1, wn = w & 1;           // 4 x 2 wave grid
    const long row0 = (long)blockIdx.x * 128;
    const long col0 = (long)blockIdx.y * 128;

    v8f acc[2][4] = {};

    for (int k0 = 0; k0 < K; k0 += 32) {
        // ---- stage A and B tiles (each 128 rows x 32 bf16), 2 chunks/thread ----
#pragma unroll
        for (int i = 0; i < 2; ++i) {
            const int id = tid * 2 + i;
            const int r = id >> 2, c8 = (id & 3) * 8;
            const __bf16* agp = A + (row0 + r) * (long)K + k0 + c8;
            const __bf16* bgp = Wt + (col0 + r) * (long)K + k0 + c8;
            ld_tile16(agp, &As[r * LDT_S + c8]);
            ld_tile16(bgp, &Bs[r * LDT_S + c8]);
            __builtin_prefetch(agp + 32, 0, 1);  // speculative, global_prefetch_b8
            __builtin_prefetch(bgp + 32, 0, 1);
        }
        async_wait0();
        __syncthreads();

        // ---- fragments: contiguous 16B LDS reads per ISA lane layouts ----
        FragBF afr[2], bfr[4];
#pragma unroll
        for (int i = 0; i < 2; ++i) {
            const __bf16* base = &As[(wm * 32 + i * 16 + li) * LDT_S + half * 8];
            afr[i].q[0] = *(const uint4*)(base);        // k = kb .. kb+7
            afr[i].q[1] = *(const uint4*)(base + 16);   // k = kb+16 .. kb+23
        }
#pragma unroll
        for (int j = 0; j < 4; ++j) {
            const __bf16* base = &Bs[(wn * 64 + j * 16 + li) * LDT_S + half * 8];
            bfr[j].q[0] = *(const uint4*)(base);        // k = kb .. kb+7
            bfr[j].q[1] = *(const uint4*)(base + 16);   // k = kb+16 .. kb+23
        }
#pragma unroll
        for (int i = 0; i < 2; ++i)
#pragma unroll
            for (int j = 0; j < 4; ++j)
                acc[i][j] = wmma_bf16(afr[i], bfr[j], acc[i][j]);
        __syncthreads();
    }

    // ---- epilogue ----
#pragma unroll
    for (int i = 0; i < 2; ++i) {
        const long mbase = row0 + wm * 32 + i * 16 + half * 8;
#pragma unroll
        for (int j = 0; j < 4; ++j) {
            const long ncol = col0 + wn * 64 + j * 16 + li;
            const float bv = bias[ncol];
#pragma unroll
            for (int r = 0; r < 8; ++r) {
                const long oidx = (mbase + r) * (long)N + ncol;
                float val = acc[i][j][r] + bv;
                if (epi == 0) {
                    ((__bf16*)out)[oidx] = (__bf16)val;
                } else if (epi == 1) {
                    ((__bf16*)out)[oidx] = (__bf16)fmaxf(val, 0.f);
                } else {
                    ((float*)out)[oidx] = resid[oidx] + val;
                }
            }
        }
    }
}

// ---------------------------------------------------------------------------
// Flash attention: one block per (q-tile of 64, head, batch); 4 waves,
// each wave owns 16 query rows. Scores and P@V via v_wmma (bf16).
// ---------------------------------------------------------------------------
#define QLS 72   // padded LDS stride (bf16), 144B rows, 16B aligned

__global__ __launch_bounds__(128)
void flash_attn_kernel(const __bf16* __restrict__ Q, const __bf16* __restrict__ Kg,
                       const __bf16* __restrict__ Vg, const int* __restrict__ mask,
                       __bf16* __restrict__ ctx) {
    __shared__ __attribute__((aligned(16))) __bf16 Qs[64 * QLS];
    __shared__ __attribute__((aligned(16))) __bf16 Ks[64 * QLS];      // [key][d]
    __shared__ __attribute__((aligned(16))) __bf16 Vt[64 * QLS];      // [d][key]
    __shared__ __attribute__((aligned(16))) __bf16 Ps[4 * 16 * QLS];  // per wave P tile

    const int tid = threadIdx.x;
    const int lane = tid & 31, w = tid >> 5;
    const int half = lane >> 4, li = lane & 15;
    const int qt = blockIdx.x, h = blockIdx.y, b = blockIdx.z;
    const long rowbase = (long)b * SEQ;
    const long qrow0 = rowbase + (long)qt * 64;
    const int colh = h * DKH;

    // ---- load Q tile (64 x 64) ----
    for (int c = 0; c < 4; ++c) {
        const int id = tid * 4 + c;
        const int r = id >> 3, c8 = (id & 7) * 8;
        ld_tile16(Q + (qrow0 + r) * (long)D_MODEL + colh + c8, &Qs[r * QLS + c8]);
    }
    async_wait0();
    __syncthreads();

    FragBF qf[2];
#pragma unroll
    for (int kc = 0; kc < 2; ++kc) {
        const __bf16* base = &Qs[(w * 16 + li) * QLS + kc * 32 + half * 8];
        qf[kc].q[0] = *(const uint4*)(base);
        qf[kc].q[1] = *(const uint4*)(base + 16);
    }

    float m_i[8], l_i[8];
    v8f ao[4] = {};
#pragma unroll
    for (int r = 0; r < 8; ++r) { m_i[r] = -3.0e38f; l_i[r] = 0.f; }

    for (int kt = 0; kt < SEQ; kt += 64) {
        // ---- stage K (row major, async) and V (transposed, manual) ----
        for (int c = 0; c < 4; ++c) {
            const int id = tid * 4 + c;
            const int r = id >> 3, c8 = (id & 7) * 8;
            ld_tile16(Kg + (rowbase + kt + r) * (long)D_MODEL + colh + c8,
                      &Ks[r * QLS + c8]);
            uint4 vv = *(const uint4*)(Vg + (rowbase + kt + r) * (long)D_MODEL + colh + c8);
            const __bf16* vp = (const __bf16*)&vv;
#pragma unroll
            for (int j = 0; j < 8; ++j)
                Vt[(c8 + j) * QLS + r] = vp[j];
        }
        async_wait0();
        __syncthreads();

        // ---- S = Q K^T (4 key sub-tiles of 16) ----
        v8f sc[4];
#pragma unroll
        for (int nt = 0; nt < 4; ++nt) {
            v8f z = {};
#pragma unroll
            for (int kc = 0; kc < 2; ++kc) {
                FragBF kf;
                const __bf16* base = &Ks[(nt * 16 + li) * QLS + kc * 32 + half * 8];
                kf.q[0] = *(const uint4*)(base);
                kf.q[1] = *(const uint4*)(base + 16);
                z = wmma_bf16(qf[kc], kf, z);
            }
            sc[nt] = z;
        }

        // ---- scale + mask (reference order: scale, then mask to -1e9) ----
#pragma unroll
        for (int nt = 0; nt < 4; ++nt) {
            const int key = kt + nt * 16 + li;
            const int mv = mask[(long)b * SEQ + key];
#pragma unroll
            for (int r = 0; r < 8; ++r) {
                float s = sc[nt][r] * 0.125f;  // 1/sqrt(64)
                sc[nt][r] = (mv == 0) ? -1.0e9f : s;
            }
        }

        // ---- online softmax (row reductions across half-wave lanes) ----
        float alpha[8];
#pragma unroll
        for (int r = 0; r < 8; ++r) {
            float mx = fmaxf(fmaxf(sc[0][r], sc[1][r]), fmaxf(sc[2][r], sc[3][r]));
#pragma unroll
            for (int d = 1; d < 16; d <<= 1) mx = fmaxf(mx, __shfl_xor(mx, d, 32));
            const float mn = fmaxf(m_i[r], mx);
            alpha[r] = __expf(m_i[r] - mn);
            m_i[r] = mn;
            float rs = 0.f;
#pragma unroll
            for (int nt = 0; nt < 4; ++nt) {
                const float p = __expf(sc[nt][r] - mn);
                sc[nt][r] = p;
                rs += p;
            }
#pragma unroll
            for (int d = 1; d < 16; d <<= 1) rs += __shfl_xor(rs, d, 32);
            l_i[r] = l_i[r] * alpha[r] + rs;
#pragma unroll
            for (int dt = 0; dt < 4; ++dt) ao[dt][r] *= alpha[r];
        }

        // ---- stage P (C-layout -> LDS -> A-layout) ----
        __bf16* pw = &Ps[w * 16 * QLS];
#pragma unroll
        for (int nt = 0; nt < 4; ++nt)
#pragma unroll
            for (int r = 0; r < 8; ++r)
                pw[(r + half * 8) * QLS + nt * 16 + li] = (__bf16)sc[nt][r];
        __syncthreads();

        FragBF pf[2];
#pragma unroll
        for (int kc = 0; kc < 2; ++kc) {
            const __bf16* base = &pw[li * QLS + kc * 32 + half * 8];
            pf[kc].q[0] = *(const uint4*)(base);
            pf[kc].q[1] = *(const uint4*)(base + 16);
        }

        // ---- O += P @ V ----
#pragma unroll
        for (int dt = 0; dt < 4; ++dt) {
#pragma unroll
            for (int kc = 0; kc < 2; ++kc) {
                FragBF vf;
                const __bf16* base = &Vt[(dt * 16 + li) * QLS + kc * 32 + half * 8];
                vf.q[0] = *(const uint4*)(base);
                vf.q[1] = *(const uint4*)(base + 16);
                ao[dt] = wmma_bf16(pf[kc], vf, ao[dt]);
            }
        }
        __syncthreads();
    }

    // ---- normalize and store ctx ----
#pragma unroll
    for (int dt = 0; dt < 4; ++dt) {
#pragma unroll
        for (int r = 0; r < 8; ++r) {
            const long row = qrow0 + w * 16 + half * 8 + r;
            const float o = ao[dt][r] / l_i[r];
            ctx[row * (long)D_MODEL + colh + dt * 16 + li] = (__bf16)o;
        }
    }
}

// ---------------------------------------------------------------------------
// Host-side orchestration
// ---------------------------------------------------------------------------
extern "C" void kernel_launch(void* const* d_in, const int* in_sizes, int n_in,
                              void* d_out, int out_size, void* d_ws, size_t ws_size,
                              hipStream_t stream) {
    (void)in_sizes; (void)n_in; (void)out_size; (void)ws_size;

    const float* x    = (const float*)d_in[0];
    const int*   mask = (const int*)d_in[1];
    const float* wq = (const float*)d_in[2];  const float* bq = (const float*)d_in[3];
    const float* wk = (const float*)d_in[4];  const float* bk = (const float*)d_in[5];
    const float* wv = (const float*)d_in[6];  const float* bv = (const float*)d_in[7];
    const float* wo = (const float*)d_in[8];  const float* bo = (const float*)d_in[9];
    const float* w1 = (const float*)d_in[10]; const float* b1 = (const float*)d_in[11];
    const float* w2 = (const float*)d_in[12]; const float* b2 = (const float*)d_in[13];
    const float* ln1_a = (const float*)d_in[14]; const float* ln1_b = (const float*)d_in[15];
    const float* ln2_a = (const float*)d_in[16]; const float* ln2_b = (const float*)d_in[17];
    float* out = (float*)d_out;

    const long MROWS = 4L * SEQ;  // 8192

    // workspace carve-up (bf16 elements)
    __bf16* wsb = (__bf16*)d_ws;
    size_t off = 0;
    auto alloc = [&](size_t n) { __bf16* p = wsb + off; off += n; return p; };
    __bf16* wqT = alloc((size_t)D_MODEL * D_MODEL);   // [N][K]
    __bf16* wkT = alloc((size_t)D_MODEL * D_MODEL);
    __bf16* wvT = alloc((size_t)D_MODEL * D_MODEL);
    __bf16* woT = alloc((size_t)D_MODEL * D_MODEL);
    __bf16* w1T = alloc((size_t)D_FF * D_MODEL);      // [3072][768]
    __bf16* w2T = alloc((size_t)D_MODEL * D_FF);      // [768][3072]
    __bf16* xn  = alloc((size_t)MROWS * D_MODEL);
    __bf16* qb  = alloc((size_t)MROWS * D_MODEL);
    __bf16* kb  = alloc((size_t)MROWS * D_MODEL);
    __bf16* vb  = alloc((size_t)MROWS * D_MODEL);
    __bf16* cb  = alloc((size_t)MROWS * D_MODEL);
    __bf16* hb  = alloc((size_t)MROWS * D_FF);

    // 1. transpose-convert weights: W[K][N] fp32 -> Wt[N][K] bf16
    dim3 g768(D_MODEL / 32, D_MODEL / 32);
    transpose_cvt_kernel<<<g768, 256, 0, stream>>>(wq, wqT, D_MODEL, D_MODEL);
    transpose_cvt_kernel<<<g768, 256, 0, stream>>>(wk, wkT, D_MODEL, D_MODEL);
    transpose_cvt_kernel<<<g768, 256, 0, stream>>>(wv, wvT, D_MODEL, D_MODEL);
    transpose_cvt_kernel<<<g768, 256, 0, stream>>>(wo, woT, D_MODEL, D_MODEL);
    transpose_cvt_kernel<<<dim3(D_MODEL / 32, D_FF / 32), 256, 0, stream>>>(w1, w1T, D_MODEL, D_FF);
    transpose_cvt_kernel<<<dim3(D_FF / 32, D_MODEL / 32), 256, 0, stream>>>(w2, w2T, D_FF, D_MODEL);

    // 2. LN1
    ln_bf16_kernel<<<dim3((unsigned)MROWS), 256, 0, stream>>>(x, ln1_a, ln1_b, xn);

    // 3. Q/K/V projections (epi 0: bf16 store)
    dim3 gproj(MROWS / 128, D_MODEL / 128);
    gemm_bf16_wmma_kernel<<<gproj, 256, 0, stream>>>(xn, wqT, bq, qb, nullptr,
                                                     (int)MROWS, D_MODEL, D_MODEL, 0);
    gemm_bf16_wmma_kernel<<<gproj, 256, 0, stream>>>(xn, wkT, bk, kb, nullptr,
                                                     (int)MROWS, D_MODEL, D_MODEL, 0);
    gemm_bf16_wmma_kernel<<<gproj, 256, 0, stream>>>(xn, wvT, bv, vb, nullptr,
                                                     (int)MROWS, D_MODEL, D_MODEL, 0);

    // 4. flash attention -> ctx (bf16)
    flash_attn_kernel<<<dim3(SEQ / 64, NHEAD, 4), 128, 0, stream>>>(qb, kb, vb, mask, cb);

    // 5. O-projection + residual (epi 2: f32, resid = x) -> d_out
    gemm_bf16_wmma_kernel<<<gproj, 256, 0, stream>>>(cb, woT, bo, out, x,
                                                     (int)MROWS, D_MODEL, D_MODEL, 2);

    // 6. LN2 on x1 (d_out)
    ln_bf16_kernel<<<dim3((unsigned)MROWS), 256, 0, stream>>>(out, ln2_a, ln2_b, xn);

    // 7. FFN1 with ReLU (epi 1)
    dim3 gff1(MROWS / 128, D_FF / 128);
    gemm_bf16_wmma_kernel<<<gff1, 256, 0, stream>>>(xn, w1T, b1, hb, nullptr,
                                                    (int)MROWS, D_FF, D_MODEL, 1);

    // 8. FFN2 + residual (epi 2, resid = out, in-place) -> d_out
    gemm_bf16_wmma_kernel<<<gproj, 256, 0, stream>>>(hb, w2T, b2, out, out,
                                                     (int)MROWS, D_MODEL, D_FF, 2);
}
